// SpatialAndChannelGlobalEnhance_53343493816801
// MI455X (gfx1250) — matrix operand order repaired and checked by reading, not compiled, not run on hardware
//
#include <hip/hip_runtime.h>

typedef __bf16 bf16_t;
typedef bf16_t v16bf __attribute__((ext_vector_type(16)));
typedef float  v8f   __attribute__((ext_vector_type(8)));
typedef unsigned int v8u  __attribute__((ext_vector_type(8)));
typedef unsigned int u32x4 __attribute__((ext_vector_type(4)));
typedef int i32x8 __attribute__((ext_vector_type(8)));
typedef int i32x4 __attribute__((ext_vector_type(4)));

#define BATCH 8
#define CH    256
#define HH    128
#define WW    128
#define HW    16384            // HH*WW
#define NTILE 64
#define TILES_PER_B (HW / NTILE)   // 256

// ---------- helpers ----------
__device__ __forceinline__ unsigned f2ord(float f) {
  unsigned u = __float_as_uint(f);
  return (u & 0x80000000u) ? ~u : (u | 0x80000000u);
}
__device__ __forceinline__ float ord2f(unsigned o) {
  unsigned u = (o & 0x80000000u) ? (o ^ 0x80000000u) : ~o;
  return __uint_as_float(u);
}
__device__ __forceinline__ unsigned short f2bf(float f) {   // RNE f32->bf16
  unsigned u = __float_as_uint(f);
  unsigned r = (u + 0x7FFFu + ((u >> 16) & 1u)) >> 16;
  return (unsigned short)r;
}
__device__ __forceinline__ float sigmoidf_(float v) { return 1.f / (1.f + __expf(-v)); }

// Issue a TDM 2D tensor load: 256x256 bf16 (stride 256) from `gsrc` to LDS offset `lds_addr`.
__device__ __forceinline__ void tdm_load_vw(const void* gsrc, unsigned lds_addr) {
  unsigned long long ga = (unsigned long long)gsrc;
  // D# group0: [1:0]=count=1, [63:32]=lds_addr, [120:64]=global_addr, [127:126]=type=2
  u32x4 g0 = { 1u, lds_addr, (unsigned)ga,
               (unsigned)((ga >> 32) & 0x01FFFFFFu) | (2u << 30) };
  // D# group1: data_size=1 (2B) @bits17:16; tensor_dim0=256 @79:48; tensor_dim1=256 @111:80;
  //            tile_dim0=256 @127:112; tile_dim1=256 @143:128; tensor_dim0_stride=256 @207:160
  i32x8 g1 = { 0x00010000, 0x01000000, 0x01000000, 0x01000000,
               0x00000100, 0x00000100, 0x00000000, 0x00000000 };
  i32x4 z4 = { 0, 0, 0, 0 };
#if __has_include(<hip/amd_detail/amd_gfx1250_TDM.h>)
  i32x8 z8 = { 0, 0, 0, 0, 0, 0, 0, 0 };
  __builtin_amdgcn_tensor_load_to_lds(g0, g1, z4, z4, z8, 0);   // clang-23 / therock: 6 args
#else
  __builtin_amdgcn_tensor_load_to_lds(g0, g1, z4, z4, 0);       // ROCm 7.2 clang-22: 5 args
#endif
}

// ---------- K0: init atomic buffers ----------
__global__ void k_init(unsigned* wmax, float* wsum, float* cfea) {
  int t = blockIdx.x * 256 + threadIdx.x;
  if (t < BATCH * WW) { wmax[t] = f2ord(-__builtin_inff()); wsum[t] = 0.f; }
  if (t < BATCH * CH) cfea[t] = 0.f;
}

// ---------- K0b: v_w f32 -> packed bf16 ----------
__global__ void k_cvt(const float* __restrict__ vw, unsigned* __restrict__ wbf) {
  int i = blockIdx.x * 256 + threadIdx.x;          // 0..32767 (pairs)
  float a = vw[2 * i], b = vw[2 * i + 1];
  wbf[i] = (unsigned)f2bf(a) | ((unsigned)f2bf(b) << 16);
}

// ---------- K1: one pass over x: k-logits, h stats, w stats ----------
__global__ __launch_bounds__(256)
void k_reduce(const float* __restrict__ x, const float* __restrict__ kw,
              const float* __restrict__ kb, float* __restrict__ klog,
              unsigned* __restrict__ wmax, float* __restrict__ wsum,
              float* __restrict__ hmax, float* __restrict__ hsum) {
  __shared__ float s_k[256], s_m[256], s_s[256];
  int b = blockIdx.x >> 7, h = blockIdx.x & 127;
  int t = threadIdx.x, w = t & 127, half = t >> 7;
  const float* xp = x + (size_t)b * CH * HW + h * WW + w;
  float ka = 0.f, mx = -__builtin_inff(), sm = 0.f;
  for (int c = half; c < CH; c += 2) {
    float v = xp[(size_t)c * HW];                  // lanes contiguous in w
    ka += kw[c] * v;
    mx = fmaxf(mx, v);
    sm += v;
  }
  s_k[t] = ka; s_m[t] = mx; s_s[t] = sm;
  __syncthreads();
  if (t < 128) {
    float kf = s_k[t] + s_k[t + 128];
    float mf = fmaxf(s_m[t], s_m[t + 128]);
    float sf = s_s[t] + s_s[t + 128];
    klog[(size_t)b * HW + h * WW + t] = kf + kb[0];
    atomicMax(&wmax[b * WW + t], f2ord(mf));       // per-column max over (c,h)
    atomicAdd(&wsum[b * WW + t], sf);
    s_m[t] = mf; s_s[t] = sf;
  }
  __syncthreads();
  for (int off = 64; off >= 1; off >>= 1) {
    if (t < off) { s_m[t] = fmaxf(s_m[t], s_m[t + off]); s_s[t] += s_s[t + off]; }
    __syncthreads();
  }
  if (t == 0) { hmax[b * HH + h] = s_m[0]; hsum[b * HH + h] = s_s[0]; }
}

// ---------- K2: per-batch softmax over spatial + h/w attention ----------
__global__ __launch_bounds__(256)
void k_softmax_att(float* __restrict__ klog,
                   const float* __restrict__ hmax, const float* __restrict__ hsum,
                   const unsigned* __restrict__ wmax, const float* __restrict__ wsum,
                   const float* __restrict__ w1, const float* __restrict__ b1,
                   const float* __restrict__ w2, const float* __restrict__ b2,
                   float* __restrict__ hatt, float* __restrict__ watt) {
  __shared__ float red[256];
  int b = blockIdx.x, t = threadIdx.x;
  float* kp = klog + (size_t)b * HW;
  float mx = -__builtin_inff();
  for (int i = t; i < HW; i += 256) mx = fmaxf(mx, kp[i]);
  red[t] = mx; __syncthreads();
  for (int off = 128; off >= 1; off >>= 1) { if (t < off) red[t] = fmaxf(red[t], red[t + off]); __syncthreads(); }
  mx = red[0]; __syncthreads();
  float sm = 0.f;
  for (int i = t; i < HW; i += 256) sm += __expf(kp[i] - mx);
  red[t] = sm; __syncthreads();
  for (int off = 128; off >= 1; off >>= 1) { if (t < off) red[t] += red[t + off]; __syncthreads(); }
  float inv = 1.f / red[0];
  for (int i = t; i < HW; i += 256) kp[i] = __expf(kp[i] - mx) * inv;
  if (t < HH) {
    float ha = hsum[b * HH + t] * (1.f / (float)(CH * WW));
    hatt[b * HH + t] = sigmoidf_(w1[0] * hmax[b * HH + t] + w1[1] * ha + b1[0]);
  } else if (t < HH + WW) {
    int w = t - HH;
    float wa = wsum[b * WW + w] * (1.f / (float)(CH * HH));
    watt[b * WW + w] = sigmoidf_(w2[0] * ord2f(wmax[b * WW + w]) + w2[1] * wa + b2[0]);
  }
}

// ---------- K3: WMMA GEMM (tmp = v_w*x + v_b) fused with both epilogues ----------
__global__ __launch_bounds__(256)
void k_gemm(const float* __restrict__ x, const unsigned* __restrict__ wbf,
            const float* __restrict__ vb, const float* __restrict__ w3,
            const float* __restrict__ b3, const float* __restrict__ knorm,
            const float* __restrict__ hatt, const float* __restrict__ watt,
            float* __restrict__ cfea, float* __restrict__ sbuf) {
  extern __shared__ char smem[];
  unsigned* lds_w = (unsigned*)smem;                    // 32768 u32: v_w bf16 [o][c]
  unsigned* lds_x = (unsigned*)(smem + 131072);         // 8192  u32: x tile bf16 [n][c]
  float*    slog  = (float*)(smem + 131072 + 32768);    // 64 floats

  int bx = blockIdx.x;
  int b  = bx >> 8;                                     // TILES_PER_B == 256
  int n0 = (bx & 255) * NTILE;
  int t  = threadIdx.x;
  int wid = t >> 5, lane = t & 31, lg = lane >> 4, l15 = lane & 15;

  // Wave 0 kicks off the TDM DMA of the 128 KB bf16 v_w panel into LDS (async,
  // tracked by TENSORcnt), overlapped with the VALU x-tile transpose/convert below.
  if (t < 32) {
    tdm_load_vw((const void*)wbf, (unsigned)(unsigned long long)(uintptr_t)smem);
  }

  // stage x tile transposed to [n][c] as packed bf16 (needs f32->bf16, so VALU path)
  {
    const float* xb = x + (size_t)b * CH * HW + n0;
    for (int i = t; i < 8192; i += 256) {
      int n = i & 63, cp = i >> 6;                      // cp = channel pair
      float a = xb[(size_t)(2 * cp) * HW + n];
      float c2 = xb[(size_t)(2 * cp + 1) * HW + n];
      lds_x[n * 128 + cp] = (unsigned)f2bf(a) | ((unsigned)f2bf(c2) << 16);
    }
  }
  if (t < NTILE) slog[t] = 0.f;
  if (t < 32) __builtin_amdgcn_s_wait_tensorcnt(0);     // wave 0 waits for the DMA
  __syncthreads();                                       // publishes LDS to all waves

  // wave 'wid' owns M-tiles {2w, 2w+1}, all 4 N-tiles
  v8f acc[2][4];
  for (int m = 0; m < 2; m++)
    for (int nt = 0; nt < 4; nt++) {
      v8f z = {0.f, 0.f, 0.f, 0.f, 0.f, 0.f, 0.f, 0.f};
      acc[m][nt] = z;
    }

  for (int kt = 0; kt < 8; kt++) {                      // K = 256 in steps of 32
    v16bf af[2];
    for (int m = 0; m < 2; m++) {
      // A (16x32 bf16): lane=M (mod 16); lanes0-15 K={0..7,16..23}, lanes16-31 K={8..15,24..31}
      int M = (wid * 2 + m) * 16 + l15;
      const uint4* p = (const uint4*)(lds_w + M * 128 + kt * 16 + lg * 4);
      uint4 lo = p[0], hi = p[2];
      v8u r; r[0]=lo.x; r[1]=lo.y; r[2]=lo.z; r[3]=lo.w; r[4]=hi.x; r[5]=hi.y; r[6]=hi.z; r[7]=hi.w;
      af[m] = __builtin_bit_cast(v16bf, r);
    }
    for (int nt = 0; nt < 4; nt++) {
      // B (32x16 bf16): lane=N (mod 16); lanes0-15 K={0..7,16..23}, lanes16-31 K={8..15,24..31}
      int N = nt * 16 + l15;
      const uint4* p = (const uint4*)(lds_x + N * 128 + kt * 16 + lg * 4);
      uint4 lo = p[0], hi = p[2];
      v8u r; r[0]=lo.x; r[1]=lo.y; r[2]=lo.z; r[3]=lo.w; r[4]=hi.x; r[5]=hi.y; r[6]=hi.z; r[7]=hi.w;
      v16bf bf = __builtin_bit_cast(v16bf, r);
      acc[0][nt] = __builtin_amdgcn_wmma_f32_16x16x32_bf16(false, af[0], false, bf, (short)0, acc[0][nt], false, false);
      acc[1][nt] = __builtin_amdgcn_wmma_f32_16x16x32_bf16(false, af[1], false, bf, (short)0, acc[1][nt], false, false);
    }
  }

  // Epilogue. C/D layout: VGPR r holds M = r (lanes 0-15) or 8+r (lanes 16-31); N = lane&15.
  const float* kp = knorm + (size_t)b * HW + n0;
  float sp[4] = {0.f, 0.f, 0.f, 0.f};
  for (int m = 0; m < 2; m++) {
    int Mb = (wid * 2 + m) * 16 + lg * 8;
    for (int r = 0; r < 8; r++) {
      int M = Mb + r;
      float vbv = vb[M], w3v = w3[M];
      float cp = 0.f;
      for (int nt = 0; nt < 4; nt++) {
        float val = acc[m][nt][r] + vbv;               // tmp with bias
        cp += val * kp[nt * 16 + l15];                 // k-weighted channel partial
        sp[nt] += w3v * val;                           // w3 . tmp partial (per column)
      }
      for (int off = 8; off; off >>= 1) cp += __shfl_xor(cp, off, 16);
      if (l15 == 0) atomicAdd(&cfea[b * CH + M], cp);
    }
  }
  for (int nt = 0; nt < 4; nt++) atomicAdd(&slog[nt * 16 + l15], sp[nt]);
  __syncthreads();

  if (t < NTILE) {
    int n = n0 + t;
    int hh = n >> 7, ww = n & 127;
    float att = hatt[b * HH + hh] + watt[b * WW + ww];
    float sl = slog[t] * att + b3[0];
    sbuf[(size_t)b * HW + n] = sigmoidf_(sl);
  }
}

// ---------- K4: out = s * channel_fea + x (float4) ----------
__global__ __launch_bounds__(256)
void k_out(const float* __restrict__ x, const float* __restrict__ sbuf,
           const float* __restrict__ cfea, float* __restrict__ out) {
  size_t i = ((size_t)blockIdx.x * 256 + threadIdx.x) * 4;
  int b = (int)(i >> 22);                              // CH*HW = 2^22 per batch
  int c = (int)((i >> 14) & 255);
  int n = (int)(i & 16383);
  float sc = cfea[b * CH + c];
  float4 xv = *(const float4*)(x + i);
  float4 sv = *(const float4*)(sbuf + ((size_t)b << 14) + n);
  float4 o;
  o.x = sv.x * sc + xv.x;
  o.y = sv.y * sc + xv.y;
  o.z = sv.z * sc + xv.z;
  o.w = sv.w * sc + xv.w;
  *(float4*)(out + i) = o;
}

extern "C" void kernel_launch(void* const* d_in, const int* in_sizes, int n_in,
                              void* d_out, int out_size, void* d_ws, size_t ws_size,
                              hipStream_t stream) {
  const float* x  = (const float*)d_in[0];
  const float* vw = (const float*)d_in[1];
  const float* vb = (const float*)d_in[2];
  const float* kw = (const float*)d_in[3];
  const float* kb = (const float*)d_in[4];
  const float* w1 = (const float*)d_in[5];
  const float* b1 = (const float*)d_in[6];
  const float* w2 = (const float*)d_in[7];
  const float* b2 = (const float*)d_in[8];
  const float* w3 = (const float*)d_in[9];
  const float* b3 = (const float*)d_in[10];
  float* out = (float*)d_out;

  char* ws = (char*)d_ws;
  float*    klog = (float*)ws;          ws += (size_t)BATCH * HW * 4;   // 512 KB (k logits -> k norm)
  float*    sbuf = (float*)ws;          ws += (size_t)BATCH * HW * 4;   // 512 KB
  unsigned* wmax = (unsigned*)ws;       ws += (size_t)BATCH * WW * 4;
  float*    wsum = (float*)ws;          ws += (size_t)BATCH * WW * 4;
  float*    hmax = (float*)ws;          ws += (size_t)BATCH * HH * 4;
  float*    hsum = (float*)ws;          ws += (size_t)BATCH * HH * 4;
  float*    hatt = (float*)ws;          ws += (size_t)BATCH * HH * 4;
  float*    watt = (float*)ws;          ws += (size_t)BATCH * WW * 4;
  float*    cfea = (float*)ws;          ws += (size_t)BATCH * CH * 4;
  unsigned* wbf  = (unsigned*)ws;       ws += (size_t)CH * CH / 2 * 4;  // v_w bf16 packed

  k_init<<<8, 256, 0, stream>>>(wmax, wsum, cfea);
  k_cvt<<<128, 256, 0, stream>>>(vw, wbf);
  k_reduce<<<BATCH * HH, 256, 0, stream>>>(x, kw, kb, klog, wmax, wsum, hmax, hsum);
  k_softmax_att<<<BATCH, 256, 0, stream>>>(klog, hmax, hsum, wmax, wsum, w1, b1, w2, b2, hatt, watt);
  k_gemm<<<BATCH * TILES_PER_B, 256, 131072 + 32768 + 256, stream>>>(
      x, wbf, vb, w3, b3, klog, hatt, watt, cfea, sbuf);
  k_out<<<32768, 256, 0, stream>>>(x, sbuf, cfea, out);
}
